// PygGraphSAGEEncoder_53764400611779
// MI455X (gfx1250) — matrix operand order repaired and checked by reading, not compiled, not run on hardware
//
#include <hip/hip_runtime.h>
#include <hip/hip_bf16.h>

typedef __attribute__((ext_vector_type(16))) _Float16 v16h;
typedef __attribute__((ext_vector_type(8)))  float    v8f;
typedef __attribute__((ext_vector_type(4)))  _Float16 v4h;

#define B_ 8
#define N_ 10000
#define E_ 160000
#define H_ 128
#define Z_ 64

// ---- workspace layout (floats / halves) -------------------------------------
// zeroed f32 region:
//   deg      : [0, N_)                                    10000 f32
//   aggsum1  : [N_, N_+B_*N_)                              80000 f32
//   poolsum  : [90000, 90000+1024)                          1024 f32
//   aggsum2  : [91024, 91024+B_*N_*H_)                  10240000 f32
// total zero region = 10331024 f32 = 41324096 bytes (64B aligned)
// f16 region:
//   h1     : B_*N_*H_ = 10240000 f16  @ byte 41324096
//   WcatB  : 8*8*32*16 = 32768 f16    @ byte 61804096
#define OFF_DEG      0
#define OFF_AGG1     (N_)
#define OFF_POOL     (N_ + B_*N_)
#define OFF_AGG2     (N_ + B_*N_ + 1024)
#define ZERO_F32     (N_ + B_*N_ + 1024 + B_*N_*H_)
#define OFF_H1_BYTES ((size_t)ZERO_F32 * 4)
#define OFF_WC_BYTES (OFF_H1_BYTES + (size_t)B_*N_*H_*2)

__device__ __forceinline__ void atomAddF(float* p, float v) {
    __hip_atomic_fetch_add(p, v, __ATOMIC_RELAXED, __HIP_MEMORY_SCOPE_AGENT);
}

// ---- zero the accumulator region --------------------------------------------
__global__ void k_zero(float* __restrict__ p, int n) {
    for (int i = blockIdx.x * blockDim.x + threadIdx.x; i < n; i += gridDim.x * blockDim.x)
        p[i] = 0.0f;
}

// ---- layer-1 edge aggregation: deg + scalar segment-sum ---------------------
__global__ void k_edge1(const float* __restrict__ x,
                        const long long* __restrict__ ei,
                        float* __restrict__ deg,
                        float* __restrict__ agg1) {
    int e = blockIdx.x * blockDim.x + threadIdx.x;
    if (e >= E_) return;
    int s = (int)ei[e];
    int d = (int)ei[E_ + e];
    atomAddF(&deg[d], 1.0f);
#pragma unroll
    for (int b = 0; b < B_; ++b)
        atomAddF(&agg1[b * N_ + d], x[b * N_ + s]);
}

// ---- h1 = relu(W1l*agg + W1r*x + b1)  (f16 out) -----------------------------
__global__ void k_h1(const float* __restrict__ x,
                     const float* __restrict__ deg,
                     const float* __restrict__ agg1,
                     const float* __restrict__ W1l,
                     const float* __restrict__ b1,
                     const float* __restrict__ W1r,
                     _Float16* __restrict__ h1) {
    int bn = blockIdx.x;              // b*N_+n
    int h  = threadIdx.x;             // 0..127
    int n  = bn % N_;
    float dg = deg[n];
    float a  = agg1[bn] / fmaxf(dg, 1.0f);
    float xv = x[bn];
    float v  = fmaxf(W1l[h] * a + W1r[h] * xv + b1[h], 0.0f);
    h1[(size_t)bn * H_ + h] = (_Float16)v;
}

// ---- layer-2 SpMM: aggsum2[b,dst,:] += h1[b,src,:] --------------------------
__global__ void k_spmm(const long long* __restrict__ ei,
                       const _Float16* __restrict__ h1,
                       float* __restrict__ agg2) {
    long long idx = (long long)blockIdx.x * blockDim.x + threadIdx.x;
    int lane = (int)(idx & 31);
    long long pair = idx >> 5;        // (e,b)
    int b = (int)(pair & 7);
    int e = (int)(pair >> 3);
    if (e >= E_) return;
    int s = (int)ei[e];
    int d = (int)ei[E_ + e];
    v4h hv = *(const v4h*)(h1 + ((size_t)b * N_ + s) * H_ + lane * 4);
    float* dp = agg2 + ((size_t)b * N_ + d) * H_ + lane * 4;
#pragma unroll
    for (int i = 0; i < 4; ++i)
        atomAddF(dp + i, (float)hv[i]);
}

// ---- pre-swizzle weights into per-lane WMMA B-fragments ---------------------
// Wcat[k][col] = k<128 ? W2l[col][k] : W2r[col][k-128]
// fragment layout: WcatB[((w*8+kstep)*32 + lane)*16 + j] = Wcat[kstep*32 + (j<8 ? 8*(lane>>4) + j : 16 + 8*(lane>>4) + (j-8))][w*16 + (lane&15)]
__global__ void k_prepw(const float* __restrict__ W2l,
                        const float* __restrict__ W2r,
                        _Float16* __restrict__ WcatB) {
    int i = blockIdx.x * blockDim.x + threadIdx.x;   // 0..32767
    if (i >= 8 * 8 * 32 * 16) return;
    int j     = i & 15;
    int lane  = (i >> 4) & 31;
    int kstep = (i >> 9) & 7;
    int w     = (i >> 12) & 7;
    int hh  = lane >> 4;
    int col = w * 16 + (lane & 15);
    int k   = kstep * 32 + (j < 8 ? 8 * hh + j : 16 + 8 * hh + (j - 8));
    float v = (k < H_) ? W2l[col * H_ + k] : W2r[col * H_ + (k - H_)];
    WcatB[i] = (_Float16)v;
}

// ---- layer-2 GEMM via WMMA + fused bias/ReLU/mean-pool ----------------------
// grid (625 row-tiles, 8 batches), block 256 (8 waves; wave w = 16-col tile)
__global__ void k_gemm2(const float* __restrict__ agg2,
                        const float* __restrict__ deg,
                        const _Float16* __restrict__ h1,
                        const _Float16* __restrict__ WcatB,
                        const float* __restrict__ b2,
                        float* __restrict__ poolsum) {
    __shared__ _Float16 Asw[8 * 32 * 16];   // A tile, fragment-swizzled (8 KB)
    const int tile = blockIdx.x;
    const int b    = blockIdx.y;
    const int t    = threadIdx.x;

    // stage: thread (r=t/16, c=t%16) supplies row r, k = c*16 .. c*16+15
    {
        int r = t >> 4, c = t & 15;
        int node = tile * 16 + r;
        float vals[16];
        if (c < 8) {
            float rdeg = 1.0f / fmaxf(deg[node], 1.0f);
            const float* p = agg2 + ((size_t)b * N_ + node) * H_ + c * 16;
#pragma unroll
            for (int i = 0; i < 16; ++i) vals[i] = p[i] * rdeg;
        } else {
            const _Float16* p = h1 + ((size_t)b * N_ + node) * H_ + (c - 8) * 16;
#pragma unroll
            for (int i = 0; i < 16; ++i) vals[i] = (float)p[i];
        }
#pragma unroll
        for (int i = 0; i < 16; ++i) {
            int k     = c * 16 + i;
            int kstep = k >> 5;
            int kk    = k & 31;
            int hh    = (kk >> 3) & 1;
            int j     = (kk & 7) + (kk >= 16 ? 8 : 0);
            int lane  = hh * 16 + r;
            Asw[(kstep * 32 + lane) * 16 + j] = (_Float16)vals[i];
        }
    }
    __syncthreads();

    const int wave = t >> 5;
    const int lane = t & 31;
    const v16h* Ap = (const v16h*)Asw;
    const v16h* Bp = (const v16h*)WcatB;

    v8f c = {};
#pragma unroll
    for (int kstep = 0; kstep < 8; ++kstep) {
        v16h a  = Ap[kstep * 32 + lane];
        v16h bb = Bp[((wave * 8 + kstep) * 32) + lane];
        c = __builtin_amdgcn_wmma_f32_16x16x32_f16(false, a, false, bb,
                                                   (short)0, c, false, false);
    }

    // epilogue: bias + relu + partial row-sum into mean-pool accumulator
    int col   = wave * 16 + (lane & 15);
    float bias = b2[col];
    float s = 0.0f;
#pragma unroll
    for (int r = 0; r < 8; ++r)
        s += fmaxf(c[r] + bias, 0.0f);
    atomAddF(&poolsum[b * H_ + col], s);
}

// ---- readout MLP: pooled -> relu(Wro1) -> Wro2 ------------------------------
__global__ void k_mlp(const float* __restrict__ poolsum,
                      const float* __restrict__ Wro1,
                      const float* __restrict__ bro1,
                      const float* __restrict__ Wro2,
                      const float* __restrict__ bro2,
                      float* __restrict__ out) {
    __shared__ float pooled[B_ * H_];
    __shared__ float hid[B_ * H_];
    int t = threadIdx.x;
    for (int i = t; i < B_ * H_; i += 256)
        pooled[i] = poolsum[i] * (1.0f / (float)N_);
    __syncthreads();
    for (int task = t; task < B_ * H_; task += 256) {
        int b = task >> 7, j = task & 127;
        float s = bro1[j];
        for (int h = 0; h < H_; ++h) s += pooled[b * H_ + h] * Wro1[j * H_ + h];
        hid[task] = fmaxf(s, 0.0f);
    }
    __syncthreads();
    for (int task = t; task < B_ * Z_; task += 256) {
        int b = task >> 6, z = task & 63;
        float s = bro2[z];
        for (int j = 0; j < H_; ++j) s += hid[b * H_ + j] * Wro2[z * H_ + j];
        out[b * Z_ + z] = s;
    }
}

extern "C" void kernel_launch(void* const* d_in, const int* in_sizes, int n_in,
                              void* d_out, int out_size, void* d_ws, size_t ws_size,
                              hipStream_t stream) {
    const float*     x    = (const float*)d_in[0];
    const long long* ei   = (const long long*)d_in[1];
    const float*     W1l  = (const float*)d_in[2];
    const float*     b1   = (const float*)d_in[3];
    const float*     W1r  = (const float*)d_in[4];
    const float*     W2l  = (const float*)d_in[5];
    const float*     b2   = (const float*)d_in[6];
    const float*     W2r  = (const float*)d_in[7];
    const float*     Wro1 = (const float*)d_in[8];
    const float*     bro1 = (const float*)d_in[9];
    const float*     Wro2 = (const float*)d_in[10];
    const float*     bro2 = (const float*)d_in[11];
    float*           out  = (float*)d_out;

    float*    wsf   = (float*)d_ws;
    float*    deg   = wsf + OFF_DEG;
    float*    agg1  = wsf + OFF_AGG1;
    float*    pool  = wsf + OFF_POOL;
    float*    agg2  = wsf + OFF_AGG2;
    _Float16* h1    = (_Float16*)((char*)d_ws + OFF_H1_BYTES);
    _Float16* WcatB = (_Float16*)((char*)d_ws + OFF_WC_BYTES);

    // 1. zero accumulators
    k_zero<<<4096, 256, 0, stream>>>(wsf, ZERO_F32);
    // 2. weight pre-swizzle for WMMA B-fragments
    k_prepw<<<128, 256, 0, stream>>>(W2l, W2r, WcatB);
    // 3. layer-1 edge aggregation
    k_edge1<<<(E_ + 255) / 256, 256, 0, stream>>>(x, ei, deg, agg1);
    // 4. h1 (f16)
    k_h1<<<B_ * N_, H_, 0, stream>>>(x, deg, agg1, W1l, b1, W1r, h1);
    // 5. layer-2 SpMM (gather + float atomics)
    k_spmm<<<(E_ * B_ * 32) / 256, 256, 0, stream>>>(ei, h1, agg2);
    // 6. layer-2 GEMM via WMMA, fused relu + mean-pool
    k_gemm2<<<dim3(N_ / 16, B_), 256, 0, stream>>>(agg2, deg, h1, WcatB, b2, pool);
    // 7. readout MLP
    k_mlp<<<1, 256, 0, stream>>>(pool, Wro1, bro1, Wro2, bro2, out);
}
